// BertSelfAttention_14680198217772
// MI455X (gfx1250) — matrix operand order repaired and checked by reading, not compiled, not run on hardware
//
#include <hip/hip_runtime.h>

// ---------------------------------------------------------------------------
// BERT self-attention for MI455X (gfx1250, wave32, WMMA f16->f32).
//
// Roofline: ~80 GFLOP, ~115 MB unavoidable HBM traffic -> bandwidth-bound.
//  * One-shot f32->f16 conversion of X and Wq/Wk/Wv (removes v_cvt from the
//    GEMM hot loop, halves L2 bytes per fragment).
//  * QKV GEMM: 64x64 tile per wave (4x4 WMMA C-frags): 8 KB of f16 fragments
//    per K-step feed 16 v_wmma -> 0.5 KB/WMMA (5x less L2 traffic than 16x64).
//  * Flash attention, 32 q-rows per wave: K/V fragments shared across two
//    M-tiles, halving K/V re-reads; online softmax keeps the 0.8 GB score
//    tensor entirely in registers.
//  * Domain mask (64 MB, largest input) read exactly once: block = (batch,
//    q-tile), its 12 waves = 12 heads share the mask tile via LDS.
//  * V stored transposed [B,H,D,S] so P*V B-fragments are contiguous loads;
//    Q pre-scaled by 1/sqrt(64); global_prefetch for the next kv chunk.
// Workspace: (4*6291456 + 3*589824) f16 = 53,870,592 bytes in d_ws.
// ---------------------------------------------------------------------------

typedef __attribute__((ext_vector_type(16))) _Float16 v16h;
typedef __attribute__((ext_vector_type(8)))  float    v8f;

namespace {

constexpr int BB = 4, SS = 2048, EE = 768, HH = 12, DD = 64;
constexpr int MTOT = BB * SS;                        // 8192 rows
constexpr size_t HSD = (size_t)BB * HH * SS * DD;    // 6291456 elems / tensor
constexpr size_t XOFF = 3 * HSD;                     // f16 X copy
constexpr size_t WOFF = 4 * HSD;                     // f16 Wq|Wk|Wv
constexpr size_t XN   = (size_t)MTOT * EE;           // 6291456
constexpr size_t WN   = (size_t)3 * EE * EE;         // 1769472

union Frag {
  v16h v;
  uint4 q[2];
  _Float16 h[16];
};

__device__ inline v8f wmma_f16(v16h a, v16h b, v8f c) {
  // D = A(16x32) * B(32x16) + C, fp32 accumulate.
  return __builtin_amdgcn_wmma_f32_16x16x32_f16(false, a, false, b,
                                                (short)0, c, false, false);
}

// xor-shuffle reductions across the 16-lane half-wave (masks 1,2,4,8 never
// cross bit 4: rows 0-7 live in lanes 0-15, rows 8-15 in lanes 16-31, which
// is exactly the 16x16 C-fragment row split).
__device__ inline float half16_max(float v) {
  v = fmaxf(v, __shfl_xor(v, 1, 32));
  v = fmaxf(v, __shfl_xor(v, 2, 32));
  v = fmaxf(v, __shfl_xor(v, 4, 32));
  v = fmaxf(v, __shfl_xor(v, 8, 32));
  return v;
}
__device__ inline float half16_sum(float v) {
  v += __shfl_xor(v, 1, 32);
  v += __shfl_xor(v, 2, 32);
  v += __shfl_xor(v, 4, 32);
  v += __shfl_xor(v, 8, 32);
  return v;
}

} // namespace

// ---------------------------------------------------------------------------
// Kernel 0: one-shot f32 -> f16 conversion of X and Wq/Wk/Wv into workspace.
// 4 elements per thread; all segment sizes are multiples of 4, so a group
// never straddles a segment boundary.
// ---------------------------------------------------------------------------
__global__ __launch_bounds__(256) void cvt_kernel(
    const float* __restrict__ X,
    const float* __restrict__ Wq, const float* __restrict__ Wk,
    const float* __restrict__ Wv, _Float16* __restrict__ ws)
{
  const size_t base = ((size_t)blockIdx.x * 256 + threadIdx.x) * 4;
  const float* src;
  _Float16* dst;
  if (base < XN) {
    src = X + base;
    dst = ws + XOFF + base;
  } else {
    const size_t off = base - XN;
    const int w = (int)(off / ((size_t)EE * EE));
    const size_t r = off % ((size_t)EE * EE);
    src = ((w == 0) ? Wq : (w == 1) ? Wk : Wv) + r;
    dst = ws + WOFF + off;
  }
  const float4 f = *(const float4*)src;
  dst[0] = (_Float16)f.x; dst[1] = (_Float16)f.y;
  dst[2] = (_Float16)f.z; dst[3] = (_Float16)f.w;
}

// ---------------------------------------------------------------------------
// Kernel A: fused QKV projection, 64x64 output tile per wave (4x4 C-frags).
// NT GEMM on pre-converted f16: per K-step, 4 B-frags (W rows) are reused by
// 4 A-frags (X rows) -> 16 v_wmma per 8 KB of fragment traffic.
// WMMA lane layouts used here (A and B share one k order):
//   A 16x32: lane<16 holds row (lane&15), K = 16*(lane>>4)..+15 (contiguous)
//   B 32x16: lane holds col (lane&15), K = 16*(lane>>4)..+15 (contiguous)
// ---------------------------------------------------------------------------
__global__ __launch_bounds__(256) void qkv_proj_kernel(
    const float* __restrict__ bq, const float* __restrict__ bk,
    const float* __restrict__ bv, _Float16* __restrict__ ws)
{
  const int lane = threadIdx.x & 31;
  const int l15  = lane & 15;
  const int hi   = lane >> 4;
  const int gw   = blockIdx.x * 8 + (threadIdx.x >> 5);

  constexpr int WPT = (MTOT / 64) * (EE / 64);   // 128*12 = 1536 tiles/tensor
  const int qkv     = gw / WPT;                  // 0=Q, 1=K, 2=V
  const int rem     = gw % WPT;
  const int rowtile = rem / (EE / 64);
  const int coltile = rem % (EE / 64);           // == head index

  const _Float16* Xh = ws + XOFF;
  const _Float16* Wm = ws + WOFF + (size_t)qkv * EE * EE;
  const float* bias  = (qkv == 0) ? bq : (qkv == 1) ? bk : bv;

  const int m0 = rowtile * 64;

  const v8f zero8 = {0.f,0.f,0.f,0.f,0.f,0.f,0.f,0.f};
  v8f acc[4][4];
  #pragma unroll
  for (int i = 0; i < 4; ++i)
    #pragma unroll
    for (int j = 0; j < 4; ++j) acc[i][j] = zero8;

  for (int k0 = 0; k0 < EE; k0 += 32) {
    Frag bf[4];
    #pragma unroll
    for (int j = 0; j < 4; ++j) {
      const _Float16* wp = Wm + (size_t)(coltile * 64 + j * 16 + l15) * EE
                              + k0 + 16 * hi;
      bf[j].q[0] = *(const uint4*)(wp);
      bf[j].q[1] = *(const uint4*)(wp + 8);
    }
    #pragma unroll
    for (int i = 0; i < 4; ++i) {
      const _Float16* xp = Xh + (size_t)(m0 + 16 * i + l15) * EE + k0;
      Frag a;
      a.q[0] = *(const uint4*)(xp + 16 * hi);
      a.q[1] = *(const uint4*)(xp + 16 * hi + 8);
      #pragma unroll
      for (int j = 0; j < 4; ++j) acc[i][j] = wmma_f16(a.v, bf[j].v, acc[i][j]);
    }
  }

  // C fragment: element r -> row (mtile + r + 8*hi), col (16*j + l15).
  const int bidx = m0 >> 11;          // m0 / 2048 (64-row tile never crosses b)
  const int head = coltile;

  if (qkv == 2) {
    // V stored transposed [B,H,D,S]: rows r contiguous in s -> packed stores.
    _Float16* dstv = ws + 2 * HSD + ((size_t)(bidx * HH + head) * DD) * SS;
    #pragma unroll
    for (int i = 0; i < 4; ++i) {
      const int srow0 = (m0 & (SS - 1)) + 16 * i;
      #pragma unroll
      for (int j = 0; j < 4; ++j) {
        const int d = j * 16 + l15;
        const float bj = bias[coltile * 64 + d];
        _Float16* col = dstv + (size_t)d * SS + srow0 + 8 * hi;
        #pragma unroll
        for (int r = 0; r < 8; ++r) col[r] = (_Float16)(acc[i][j][r] + bj);
      }
    }
  } else {
    // Q (pre-scaled by 1/sqrt(D)) and K stored [B,H,S,D].
    const float scale = (qkv == 0) ? 0.125f : 1.0f;
    _Float16* dst = ws + (size_t)qkv * HSD
                       + ((size_t)(bidx * HH + head) * SS) * DD;
    #pragma unroll
    for (int i = 0; i < 4; ++i) {
      const int srow0 = (m0 & (SS - 1)) + 16 * i;
      #pragma unroll
      for (int j = 0; j < 4; ++j) {
        const int d = j * 16 + l15;
        const float bj = bias[coltile * 64 + d];
        #pragma unroll
        for (int r = 0; r < 8; ++r) {
          const int s = srow0 + r + 8 * hi;
          dst[(size_t)s * DD + d] = (_Float16)((acc[i][j][r] + bj) * scale);
        }
      }
    }
  }
}

// ---------------------------------------------------------------------------
// Kernel B: flash attention. Block = (batch, 32-row q-tile), 12 waves = heads.
// Mask tile (att + domain) staged once per block in LDS, shared by all heads.
// Each wave owns two 16-row M-tiles so K/V fragments are reused twice.
// ---------------------------------------------------------------------------
__global__ __launch_bounds__(384) void attn_kernel(
    const _Float16* __restrict__ ws,
    const float* __restrict__ attm,   // [B,1,1,S]
    const float* __restrict__ domm,   // [B,1,S,S]
    float* __restrict__ out)          // [B,S,E]
{
  __shared__ __align__(16) float    maskT[32 * 64];         // 8 KB
  __shared__ __align__(16) _Float16 pst[HH * 2 * 512];      // 24 KB P staging

  const int tid  = threadIdx.x;
  const int lane = tid & 31;
  const int wave = tid >> 5;          // head 0..11
  const int l15  = lane & 15;
  const int hi   = lane >> 4;

  const int b  = blockIdx.x >> 6;     // / (S/32)
  const int qt = blockIdx.x & 63;
  const int q0 = qt * 32;

  const _Float16* Qh = ws +           ((size_t)(b * HH + wave) * SS) * DD;
  const _Float16* Kh = ws + HSD +     ((size_t)(b * HH + wave) * SS) * DD;
  const _Float16* Vt = ws + 2 * HSD + ((size_t)(b * HH + wave) * DD) * SS;

  // Preload Q A-fragments: 2 M-tiles x 2 d-chunks, held all kernel.
  Frag qa[2][2];
  #pragma unroll
  for (int t = 0; t < 2; ++t) {
    const _Float16* qrow = Qh + (size_t)(q0 + 16 * t + l15) * DD;
    qa[t][0].q[0] = *(const uint4*)(qrow + 16 * hi);
    qa[t][0].q[1] = *(const uint4*)(qrow + 16 * hi + 8);
    qa[t][1].q[0] = *(const uint4*)(qrow + 32 + 16 * hi);
    qa[t][1].q[1] = *(const uint4*)(qrow + 32 + 16 * hi + 8);
  }

  const v8f zero8 = {0.f,0.f,0.f,0.f,0.f,0.f,0.f,0.f};
  v8f acc[2][4];
  float mrow[2][8], lrow[2][8];
  #pragma unroll
  for (int t = 0; t < 2; ++t) {
    #pragma unroll
    for (int j = 0; j < 4; ++j) acc[t][j] = zero8;
    #pragma unroll
    for (int r = 0; r < 8; ++r) { mrow[t][r] = -1e30f; lrow[t][r] = 0.f; }
  }

  for (int kv0 = 0; kv0 < SS; kv0 += 64) {
    __syncthreads();
    // Cooperative: combined mask tile, read ONCE for all 12 heads.
    for (int idx = tid; idx < 2048; idx += 384) {
      const int rw = idx >> 6, c = idx & 63;
      maskT[idx] = attm[(size_t)b * SS + kv0 + c] +
                   domm[((size_t)b * SS + q0 + rw) * SS + kv0 + c];
    }
    __syncthreads();

    // Prefetch next kv chunk's K rows and V columns (global_prefetch_b8).
    if (kv0 + 64 < SS) {
      __builtin_prefetch(Kh + (size_t)(kv0 + 64 + lane) * DD, 0, 2);
      __builtin_prefetch(Kh + (size_t)(kv0 + 96 + lane) * DD, 0, 2);
      __builtin_prefetch(Vt + (size_t)lane * SS + kv0 + 64, 0, 2);
      __builtin_prefetch(Vt + (size_t)(lane + 32) * SS + kv0 + 64, 0, 2);
    }

    #pragma unroll
    for (int sub = 0; sub < 2; ++sub) {
      const int n0 = kv0 + sub * 32;
      v8f s[2][2];

      // Scores, cols n0..n0+15: K B-frags shared by both M-tiles.
      {
        const _Float16* kr = Kh + (size_t)(n0 + l15) * DD;
        Frag kb0, kb1;
        kb0.q[0] = *(const uint4*)(kr + 16 * hi);
        kb0.q[1] = *(const uint4*)(kr + 16 * hi + 8);
        kb1.q[0] = *(const uint4*)(kr + 32 + 16 * hi);
        kb1.q[1] = *(const uint4*)(kr + 32 + 16 * hi + 8);
        #pragma unroll
        for (int t = 0; t < 2; ++t) {
          s[t][0] = wmma_f16(qa[t][0].v, kb0.v, zero8);
          s[t][0] = wmma_f16(qa[t][1].v, kb1.v, s[t][0]);
        }
      }
      // Scores, cols n0+16..n0+31.
      {
        const _Float16* kr = Kh + (size_t)(n0 + 16 + l15) * DD;
        Frag kc0, kc1;
        kc0.q[0] = *(const uint4*)(kr + 16 * hi);
        kc0.q[1] = *(const uint4*)(kr + 16 * hi + 8);
        kc1.q[0] = *(const uint4*)(kr + 32 + 16 * hi);
        kc1.q[1] = *(const uint4*)(kr + 32 + 16 * hi + 8);
        #pragma unroll
        for (int t = 0; t < 2; ++t) {
          s[t][1] = wmma_f16(qa[t][0].v, kc0.v, zero8);
          s[t][1] = wmma_f16(qa[t][1].v, kc1.v, s[t][1]);
        }
      }

      // Masks + online softmax + P staging (per M-tile).
      const int crel = sub * 32 + l15;
      #pragma unroll
      for (int t = 0; t < 2; ++t) {
        _Float16* pw = pst + (wave * 2 + t) * 512;
        float corr[8];
        #pragma unroll
        for (int r = 0; r < 8; ++r) {
          const int rw = 16 * t + r + 8 * hi;
          s[t][0][r] += maskT[rw * 64 + crel];
          s[t][1][r] += maskT[rw * 64 + crel + 16];
        }
        #pragma unroll
        for (int r = 0; r < 8; ++r) {
          const float tmax = half16_max(fmaxf(s[t][0][r], s[t][1][r]));
          const float mn   = fmaxf(mrow[t][r], tmax);
          corr[r] = __expf(mrow[t][r] - mn);
          const float p0 = __expf(s[t][0][r] - mn);
          const float p1 = __expf(s[t][1][r] - mn);
          s[t][0][r] = p0; s[t][1][r] = p1;
          lrow[t][r] = lrow[t][r] * corr[r] + half16_sum(p0 + p1);
          mrow[t][r] = mn;
        }
        #pragma unroll
        for (int j = 0; j < 4; ++j)
          #pragma unroll
          for (int r = 0; r < 8; ++r) acc[t][j][r] *= corr[r];
        // C-layout -> A-layout via wave-local LDS (same-wave DS is in-order).
        #pragma unroll
        for (int r = 0; r < 8; ++r) {
          const int rw = r + 8 * hi;
          pw[rw * 32 + l15]      = (_Float16)s[t][0][r];
          pw[rw * 32 + 16 + l15] = (_Float16)s[t][1][r];
        }
      }

      Frag pa[2];
      #pragma unroll
      for (int t = 0; t < 2; ++t) {
        const _Float16* pw = pst + (wave * 2 + t) * 512;
        pa[t].q[0] = *(const uint4*)(pw + l15 * 32 + 16 * hi);
        pa[t].q[1] = *(const uint4*)(pw + l15 * 32 + 16 * hi + 8);
      }

      // O += P * V : V B-fragment (transposed V, kv contiguous per lane) is
      // loaded once and consumed by BOTH M-tiles.
      #pragma unroll
      for (int j = 0; j < 4; ++j) {
        const _Float16* vrow = Vt + (size_t)(j * 16 + l15) * SS + n0 + 16 * hi;
        Frag vb;
        vb.q[0] = *(const uint4*)(vrow);
        vb.q[1] = *(const uint4*)(vrow + 8);
        acc[0][j] = wmma_f16(pa[0].v, vb.v, acc[0][j]);
        acc[1][j] = wmma_f16(pa[1].v, vb.v, acc[1][j]);
      }
    }
  }

  // Normalize and store fp32 output [B,S,E], E-col = head*64 + d.
  #pragma unroll
  for (int t = 0; t < 2; ++t)
    #pragma unroll
    for (int j = 0; j < 4; ++j) {
      const int d = j * 16 + l15;
      #pragma unroll
      for (int r = 0; r < 8; ++r) {
        const int srow = q0 + 16 * t + r + 8 * hi;
        out[((size_t)b * SS + srow) * EE + wave * 64 + d] =
            acc[t][j][r] / lrow[t][r];
      }
    }
}

// ---------------------------------------------------------------------------
extern "C" void kernel_launch(void* const* d_in, const int* in_sizes, int n_in,
                              void* d_out, int out_size, void* d_ws, size_t ws_size,
                              hipStream_t stream) {
  (void)in_sizes; (void)n_in; (void)out_size; (void)ws_size;
  const float* X    = (const float*)d_in[0];
  const float* attm = (const float*)d_in[1];
  const float* domm = (const float*)d_in[2];
  const float* Wq   = (const float*)d_in[3];
  const float* bq   = (const float*)d_in[4];
  const float* Wk   = (const float*)d_in[5];
  const float* bk   = (const float*)d_in[6];
  const float* Wv   = (const float*)d_in[7];
  const float* bv   = (const float*)d_in[8];
  float*    out = (float*)d_out;
  _Float16* ws  = (_Float16*)d_ws;   // needs 53,870,592 bytes

  // (XN + WN) / 4 elems-per-thread / 256 threads = 7872 blocks (exact).
  cvt_kernel<<<dim3(7872), dim3(256), 0, stream>>>(X, Wq, Wk, Wv, ws);
  // 3 tensors * 128 row-tiles * 12 col-tiles = 4608 waves / 8 per block.
  qkv_proj_kernel<<<dim3(576), dim3(256), 0, stream>>>(bq, bk, bv, ws);
  // One block per (batch, q-tile32): 4*64 = 256 blocks, 12 waves = 12 heads.
  attn_kernel<<<dim3(256), dim3(384), 0, stream>>>(ws, attm, domm, out);
}
